// InnerProductWithWeightsAffinity_1503238554110
// MI455X (gfx1250) — matrix operand order repaired and checked by reading, not compiled, not run on hardware
//
#include <hip/hip_runtime.h>
#include <hip/hip_bf16.h>

typedef __attribute__((ext_vector_type(16))) _Float16 v16h;
typedef __attribute__((ext_vector_type(8)))  float    v8f;

// ---------------------------------------------------------------------------
// Kernel 1: coeff[b,e] = tanh( sum_k weights[b,k] * A_w[e,k] + A_b[e] )
//   grid = 8 blocks (one per batch), block = 128 threads (one per e)
// ---------------------------------------------------------------------------
__global__ __launch_bounds__(128) void ipwa_coeff_kernel(
    const float* __restrict__ W,    // (8,1024)
    const float* __restrict__ Aw,   // (128,1024)
    const float* __restrict__ Ab,   // (128)
    float* __restrict__ coeff)      // (8,128)
{
    const int b = blockIdx.x;
    const int e = threadIdx.x;
    const float4* wr = (const float4*)(W + b * 1024);
    const float4* ar = (const float4*)(Aw + e * 1024);
    float acc = 0.0f;
#pragma unroll 4
    for (int k = 0; k < 256; ++k) {
        float4 w = wr[k];
        float4 a = ar[k];
        acc += w.x * a.x + w.y * a.y + w.z * a.z + w.w * a.w;
    }
    coeff[b * 128 + e] = tanhf(acc + Ab[e]);
}

// ---------------------------------------------------------------------------
// Kernel 2: projection GEMMs via v_wmma_f32_16x16x32_f16
//   px[b,n,e] = sum_d (X[b,n,d]*coeff[b,d]) * lin_w[e, d]
//   py[b,m,e] = sum_d  Y[b,m,d]            * lin_w[e, 128+d]
//   grid = 2 * 8 * 16 = 256 blocks, block = 256 threads = 8 waves.
//   Each wave computes one 16(M) x 16(E) f32 tile, K-loop 128 = 4 x 32.
// ---------------------------------------------------------------------------
__global__ __launch_bounds__(256) void ipwa_proj_kernel(
    const float* __restrict__ X,      // (8,256,128)
    const float* __restrict__ Y,      // (8,256,128)
    const float* __restrict__ coeff,  // (8,128)
    const float* __restrict__ lin_w,  // (128,256)
    float* __restrict__ px,           // (8,256,128)
    float* __restrict__ py)           // (8,256,128)
{
    const int bid   = blockIdx.x;
    const int which = bid >> 7;        // 0 = px, 1 = py
    const int rem   = bid & 127;
    const int b     = rem >> 4;
    const int n0    = (rem & 15) << 4;

    const int wave = threadIdx.x >> 5;
    const int lane = threadIdx.x & 31;
    const int g    = lane >> 4;        // half-wave group
    const int mr   = lane & 15;        // row (A/D) or column (B/D) index
    const int e0   = wave << 4;

    // A-matrix source row: X (scaled by coeff) or Y, row n0+mr of batch b.
    const float* arow = (which ? Y : X) + ((b << 8) + n0 + mr) * 128;
    const float* cf   = coeff + (b << 7);
    // B-matrix: B[K][N] = lin_w[e0+N][woff+K]; lane mr = N -> one lin_w row.
    const float* wrow = lin_w + (e0 + mr) * 256 + (which ? 128 : 0);

    v8f acc = {};
#pragma unroll
    for (int kk = 0; kk < 128; kk += 32) {
        // ---- A fragment: lane holds K = kk+8g+{0..7} and kk+16+8g+{0..7}
        const int ka = kk + g * 8;
        float4 x0 = *(const float4*)(arow + ka);
        float4 x1 = *(const float4*)(arow + ka + 4);
        float4 x2 = *(const float4*)(arow + ka + 16);
        float4 x3 = *(const float4*)(arow + ka + 20);
        if (!which) {
            float4 c0 = *(const float4*)(cf + ka);
            float4 c1 = *(const float4*)(cf + ka + 4);
            float4 c2 = *(const float4*)(cf + ka + 16);
            float4 c3 = *(const float4*)(cf + ka + 20);
            x0.x *= c0.x; x0.y *= c0.y; x0.z *= c0.z; x0.w *= c0.w;
            x1.x *= c1.x; x1.y *= c1.y; x1.z *= c1.z; x1.w *= c1.w;
            x2.x *= c2.x; x2.y *= c2.y; x2.z *= c2.z; x2.w *= c2.w;
            x3.x *= c3.x; x3.y *= c3.y; x3.z *= c3.z; x3.w *= c3.w;
        }
        v16h av;
        av[0]  = (_Float16)x0.x; av[1]  = (_Float16)x0.y;
        av[2]  = (_Float16)x0.z; av[3]  = (_Float16)x0.w;
        av[4]  = (_Float16)x1.x; av[5]  = (_Float16)x1.y;
        av[6]  = (_Float16)x1.z; av[7]  = (_Float16)x1.w;
        av[8]  = (_Float16)x2.x; av[9]  = (_Float16)x2.y;
        av[10] = (_Float16)x2.z; av[11] = (_Float16)x2.w;
        av[12] = (_Float16)x3.x; av[13] = (_Float16)x3.y;
        av[14] = (_Float16)x3.z; av[15] = (_Float16)x3.w;

        // ---- B fragment: lane holds K = kk+8g+{0..7} and kk+16+8g+{0..7} of lin_w row e0+mr
        const int kb = kk + g * 8;
        float4 w0 = *(const float4*)(wrow + kb);
        float4 w1 = *(const float4*)(wrow + kb + 4);
        float4 w2 = *(const float4*)(wrow + kb + 16);
        float4 w3 = *(const float4*)(wrow + kb + 20);
        v16h bv;
        bv[0]  = (_Float16)w0.x; bv[1]  = (_Float16)w0.y;
        bv[2]  = (_Float16)w0.z; bv[3]  = (_Float16)w0.w;
        bv[4]  = (_Float16)w1.x; bv[5]  = (_Float16)w1.y;
        bv[6]  = (_Float16)w1.z; bv[7]  = (_Float16)w1.w;
        bv[8]  = (_Float16)w2.x; bv[9]  = (_Float16)w2.y;
        bv[10] = (_Float16)w2.z; bv[11] = (_Float16)w2.w;
        bv[12] = (_Float16)w3.x; bv[13] = (_Float16)w3.y;
        bv[14] = (_Float16)w3.z; bv[15] = (_Float16)w3.w;

        acc = __builtin_amdgcn_wmma_f32_16x16x32_f16(
            /*neg_a=*/false, av, /*neg_b=*/false, bv,
            /*c_mod=*/(short)0, acc, /*reuse_a=*/false, /*reuse_b=*/false);
    }

    // ---- D tile store: element r -> row M = r + 8g, col N = mr
    float* dst = (which ? py : px) + ((b << 8) + n0) * 128 + e0 + mr;
#pragma unroll
    for (int r = 0; r < 8; ++r)
        dst[(r + 8 * g) * 128] = acc[r];
}

// ---------------------------------------------------------------------------
// Kernel 3: epilogue
//   out[b,n,m] = sum_d relu(px[b,n,d] + py[b,m,d] + lin_b[d]) * out_w[d] + out_b
//   grid = 8 * 8 * 8 = 512 blocks; each block does a 32n x 32m tile via LDS.
//   256 threads, 4 outputs each (same m, 4 consecutive n -> shared sy read).
// ---------------------------------------------------------------------------
__global__ __launch_bounds__(256) void ipwa_epilogue_kernel(
    const float* __restrict__ px,     // (8,256,128)
    const float* __restrict__ py,     // (8,256,128)
    const float* __restrict__ lin_b,  // (128)
    const float* __restrict__ out_w,  // (128)
    const float* __restrict__ out_b,  // (1)
    float* __restrict__ out)          // (8,256,256)
{
    __shared__ float sx[32][132];   // px tile (padded rows: stride 132)
    __shared__ float sy[32][132];   // py tile + lin_b
    __shared__ float sow[128];

    const int bid = blockIdx.x;
    const int b   = bid >> 6;
    const int rem = bid & 63;
    const int n0  = (rem >> 3) << 5;
    const int m0  = (rem & 7) << 5;
    const int tid = threadIdx.x;

    if (tid < 128) sow[tid] = out_w[tid];

    const float* pxb = px + ((b << 8) + n0) * 128;
    const float* pyb = py + ((b << 8) + m0) * 128;
#pragma unroll
    for (int i = 0; i < 4; ++i) {
        int idx = tid + i * 256;         // float4 index, 32 rows * 32 f4/row
        int row = idx >> 5;
        int col = (idx & 31) << 2;
        float4 v  = *(const float4*)(pxb + row * 128 + col);
        sx[row][col + 0] = v.x; sx[row][col + 1] = v.y;
        sx[row][col + 2] = v.z; sx[row][col + 3] = v.w;
        float4 u  = *(const float4*)(pyb + row * 128 + col);
        float4 lb = *(const float4*)(lin_b + col);
        sy[row][col + 0] = u.x + lb.x; sy[row][col + 1] = u.y + lb.y;
        sy[row][col + 2] = u.z + lb.z; sy[row][col + 3] = u.w + lb.w;
    }
    __syncthreads();

    const int m  = tid & 31;
    const int nb = (tid >> 5) << 2;
    float a0 = 0.f, a1 = 0.f, a2 = 0.f, a3 = 0.f;
#pragma unroll 4
    for (int d = 0; d < 128; ++d) {
        float yv = sy[m][d];
        float ow = sow[d];
        a0 += fmaxf(sx[nb + 0][d] + yv, 0.f) * ow;
        a1 += fmaxf(sx[nb + 1][d] + yv, 0.f) * ow;
        a2 += fmaxf(sx[nb + 2][d] + yv, 0.f) * ow;
        a3 += fmaxf(sx[nb + 3][d] + yv, 0.f) * ow;
    }
    const float ob = out_b[0];
    float* o = out + ((b << 8) + n0 + nb) * 256 + m0 + m;
    o[0]       = a0 + ob;
    o[256]     = a1 + ob;
    o[2 * 256] = a2 + ob;
    o[3 * 256] = a3 + ob;
}

// ---------------------------------------------------------------------------
extern "C" void kernel_launch(void* const* d_in, const int* in_sizes, int n_in,
                              void* d_out, int out_size, void* d_ws, size_t ws_size,
                              hipStream_t stream) {
    (void)in_sizes; (void)n_in; (void)out_size; (void)ws_size;

    const float* X       = (const float*)d_in[0];   // (8,256,128)
    const float* Y       = (const float*)d_in[1];   // (8,256,128)
    const float* weights = (const float*)d_in[2];   // (8,1024)
    const float* A_w     = (const float*)d_in[3];   // (128,1024)
    const float* A_b     = (const float*)d_in[4];   // (128)
    const float* lin_w   = (const float*)d_in[5];   // (128,256)
    const float* lin_b   = (const float*)d_in[6];   // (128)
    const float* out_w   = (const float*)d_in[7];   // (1,128)
    const float* out_b   = (const float*)d_in[8];   // (1)
    float*       out     = (float*)d_out;           // (8,256,256)

    float* coeff = (float*)d_ws;                    // 8*128
    float* px    = coeff + 1024;                    // 8*256*128
    float* py    = px + 8 * 256 * 128;              // 8*256*128

    ipwa_coeff_kernel<<<8, 128, 0, stream>>>(weights, A_w, A_b, coeff);
    ipwa_proj_kernel<<<256, 256, 0, stream>>>(X, Y, coeff, lin_w, px, py);
    ipwa_epilogue_kernel<<<512, 256, 0, stream>>>(px, py, lin_b, out_w, out_b, out);
}
